// Graph_Encoder_Norm_Pooling_32212254720638
// MI455X (gfx1250) — compile-verified, matrix operand order checked by reading, not run on hardware
//
#include <hip/hip_runtime.h>
#include <hip/hip_bf16.h>

// GAT encoder: h = leaky(x@W_in + b_in); 4x { GATConv -> LayerNorm -> LeakyReLU }
// N=100000 nodes, E=1600000 edges, D=128. fp32 in/out, f16 WMMA GEMMs (f32 accum).

typedef __attribute__((ext_vector_type(16))) _Float16 v16h;
typedef __attribute__((ext_vector_type(8)))  _Float16 v8h;
typedef __attribute__((ext_vector_type(8)))  float    v8f;

#define DIM 128
#define NLAYERS 4

// ---------------------------------------------------------------------------
// Pack weight matrices (f32 row-major [K=128][N=128]) into f16 B-fragments in
// the CDNA5 WMMA lane layout: fragment index = ((w*8 + ntile)*4 + kstep)*32 + lane,
// 16 halves per fragment. Lane l -> column N = ntile*16 + (l%16);
// element i -> K = kstep*32 + base + (i&7) + (i>=8 ? 16 : 0), base = (l<16)?0:8.
// ---------------------------------------------------------------------------
__global__ void pack_w_kernel(const float* __restrict__ W_in,
                              const float* __restrict__ Wl,
                              _Float16* __restrict__ Wpk, int total) {
  int tid = blockIdx.x * blockDim.x + threadIdx.x;
  if (tid >= total) return;
  int w  = tid >> 14;          // which weight matrix (0..4), 16384 halves each
  int r  = tid & 16383;
  int i  = r & 15;             // element within fragment
  int l  = (r >> 4) & 31;      // lane
  int ks = (r >> 9) & 3;       // k-step
  int nt = (r >> 11) & 7;      // n tile
  int n    = nt * 16 + (l & 15);
  int base = (l < 16) ? 0 : 8;
  int k    = ks * 32 + base + (i & 7) + ((i & 8) ? 16 : 0);
  const float* Wsrc = (w == 0) ? W_in : (Wl + (size_t)(w - 1) * DIM * DIM);
  Wpk[tid] = (_Float16)Wsrc[k * DIM + n];
}

// ---------------------------------------------------------------------------
// GEMM: C[m0:m0+16, :] = A[m0:m0+16, :] @ W   (A f32 [N,128], W packed f16)
// 8 waves per block; wave w computes the 16x16 tile at columns w*16.
// A tile staged in LDS as f16; 4 x v_wmma_f32_16x16x32_f16 per wave.
// Optional epilogue: +bias, leaky(0.01).
// ---------------------------------------------------------------------------
__global__ __launch_bounds__(256) void gemm_kernel(
    const float* __restrict__ A,
    const _Float16* __restrict__ Wpk, int w_idx,
    float* __restrict__ Cout,
    const float* __restrict__ bias, int do_leaky) {
  __shared__ _Float16 Atile[16 * DIM];   // 4 KB
  const int t  = threadIdx.x;
  const int m0 = blockIdx.x * 16;

  // Stage A tile: 256 threads x 8 contiguous floats -> f16 LDS
  {
    int row = t >> 4;
    int col = (t & 15) * 8;
    const float* ap = A + (size_t)(m0 + row) * DIM + col;
    float4 f0 = *(const float4*)ap;
    float4 f1 = *(const float4*)(ap + 4);
    v8h hv;
    hv[0] = (_Float16)f0.x; hv[1] = (_Float16)f0.y;
    hv[2] = (_Float16)f0.z; hv[3] = (_Float16)f0.w;
    hv[4] = (_Float16)f1.x; hv[5] = (_Float16)f1.y;
    hv[6] = (_Float16)f1.z; hv[7] = (_Float16)f1.w;
    *(v8h*)&Atile[row * DIM + col] = hv;
  }
  __syncthreads();

  const int wid  = t >> 5;        // n-tile (0..7)
  const int lane = t & 31;
  const int mrow  = lane & 15;
  const int kbase = (lane < 16) ? 0 : 8;
  const v16h* __restrict__ Bfrag =
      (const v16h*)Wpk + (size_t)w_idx * 1024 + wid * 4 * 32 + lane;

  v8f c = {};
  #pragma unroll
  for (int ks = 0; ks < 4; ++ks) {
    v8h alo = *(const v8h*)&Atile[mrow * DIM + ks * 32 + kbase];
    v8h ahi = *(const v8h*)&Atile[mrow * DIM + ks * 32 + kbase + 16];
    v16h a;
    #pragma unroll
    for (int i = 0; i < 8; ++i) { a[i] = alo[i]; a[i + 8] = ahi[i]; }
    v16h b = Bfrag[ks * 32];
    c = __builtin_amdgcn_wmma_f32_16x16x32_f16(
        /*neg_a=*/false, a, /*neg_b=*/false, b,
        /*c_mod=*/(short)0, c, /*reuse_a=*/false, /*reuse_b=*/false);
  }

  // C/D layout: lane l holds column N=l%16; VGPR r -> row M=r (+8 if lane>=16)
  const int col  = wid * 16 + (lane & 15);
  const int moff = (lane < 16) ? 0 : 8;
  const float bv = bias ? bias[col] : 0.0f;
  #pragma unroll
  for (int r = 0; r < 8; ++r) {
    float v = c[r] + bv;
    if (do_leaky) v = (v > 0.0f) ? v : 0.01f * v;
    Cout[(size_t)(m0 + moff + r) * DIM + col] = v;
  }
}

// ---------------------------------------------------------------------------
// Per-node attention logits: alpha_src[n] = h[n].a_src ; alpha_dst[n] = h[n].a_dst
// One wave per node (32 lanes x float4 = 128).
// ---------------------------------------------------------------------------
__global__ __launch_bounds__(256) void alphas_kernel(
    const float* __restrict__ H, const float* __restrict__ a_s,
    const float* __restrict__ a_d, float* __restrict__ out_s,
    float* __restrict__ out_d, int nnodes) {
  int wid = threadIdx.x >> 5, lane = threadIdx.x & 31;
  int node = blockIdx.x * 8 + wid;
  if (node >= nnodes) return;
  float4 h4 = *(const float4*)(H + (size_t)node * DIM + lane * 4);
  float4 s4 = *(const float4*)(a_s + lane * 4);
  float4 d4 = *(const float4*)(a_d + lane * 4);
  float ps = h4.x * s4.x + h4.y * s4.y + h4.z * s4.z + h4.w * s4.w;
  float pd = h4.x * d4.x + h4.y * d4.y + h4.z * d4.z + h4.w * d4.w;
  #pragma unroll
  for (int off = 16; off >= 1; off >>= 1) {
    ps += __shfl_xor(ps, off, 32);
    pd += __shfl_xor(pd, off, 32);
  }
  if (lane == 0) { out_s[node] = ps; out_d[node] = pd; }
}

// init accumulator to bias; init per-node max/denom
__global__ void init_kernel(float* __restrict__ acc, const float* __restrict__ bias,
                            float* __restrict__ nmax, float* __restrict__ ndenom,
                            int nnodes) {
  int idx = blockIdx.x * blockDim.x + threadIdx.x;
  if (idx < nnodes * DIM) acc[idx] = bias[idx & (DIM - 1)];
  if (idx < nnodes) { nmax[idx] = -INFINITY; ndenom[idx] = 0.0f; }
}

// float atomic max via sign-split integer atomics (monotone IEEE bit trick)
__device__ inline void atomic_max_f32(float* addr, float val) {
  if (val >= 0.0f)
    atomicMax((int*)addr, __float_as_int(val));
  else
    atomicMin((unsigned int*)addr, __float_as_uint(val));
}

// e = leaky(asrc[src]+adst[dst], 0.2); segment max over dst
__global__ void edge_max_kernel(const int* __restrict__ src, const int* __restrict__ dst,
                                const float* __restrict__ as, const float* __restrict__ ad,
                                float* __restrict__ ebuf, float* __restrict__ nmax, int ne) {
  int e = blockIdx.x * blockDim.x + threadIdx.x;
  if (e >= ne) return;
  int d = dst[e];
  float v = as[src[e]] + ad[d];
  v = (v > 0.0f) ? v : 0.2f * v;
  ebuf[e] = v;
  atomic_max_f32(&nmax[d], v);
}

// ex = exp(e - max[dst]); segment sum over dst
__global__ void edge_exp_kernel(const int* __restrict__ dst, const float* __restrict__ nmax,
                                float* __restrict__ ebuf, float* __restrict__ ndenom, int ne) {
  int e = blockIdx.x * blockDim.x + threadIdx.x;
  if (e >= ne) return;
  int d = dst[e];
  float ex = __expf(ebuf[e] - nmax[d]);
  ebuf[e] = ex;
  atomicAdd(&ndenom[d], ex);
}

// acc[dst] += (ex/denom[dst]) * h[src]   — one wave per edge, float4 per lane
__global__ __launch_bounds__(256) void scatter_kernel(
    const int* __restrict__ src, const int* __restrict__ dst,
    const float* __restrict__ ebuf, const float* __restrict__ ndenom,
    const float* __restrict__ G, float* __restrict__ acc, int ne) {
  int wid = threadIdx.x >> 5, lane = threadIdx.x & 31;
  int e = blockIdx.x * 8 + wid;
  if (e >= ne) return;
  int s = src[e], d = dst[e];
  float alpha = ebuf[e] / (ndenom[d] + 1e-16f);
  float4 g4 = *(const float4*)(G + (size_t)s * DIM + lane * 4);
  float* op = acc + (size_t)d * DIM + lane * 4;
  atomicAdd(op + 0, alpha * g4.x);
  atomicAdd(op + 1, alpha * g4.y);
  atomicAdd(op + 2, alpha * g4.z);
  atomicAdd(op + 3, alpha * g4.w);
}

// LayerNorm(D=128) + leaky(0.01) — one wave per node, in-place safe
__global__ __launch_bounds__(256) void ln_kernel(
    const float* __restrict__ acc, const float* __restrict__ gamma,
    const float* __restrict__ beta, float* __restrict__ out, int nnodes) {
  int wid = threadIdx.x >> 5, lane = threadIdx.x & 31;
  int node = blockIdx.x * 8 + wid;
  if (node >= nnodes) return;
  float4 v = *(const float4*)(acc + (size_t)node * DIM + lane * 4);
  float s = v.x + v.y + v.z + v.w;
  #pragma unroll
  for (int off = 16; off >= 1; off >>= 1) s += __shfl_xor(s, off, 32);
  float mu = s * (1.0f / 128.0f);
  float dx = v.x - mu, dy = v.y - mu, dz = v.z - mu, dw = v.w - mu;
  float q = dx * dx + dy * dy + dz * dz + dw * dw;
  #pragma unroll
  for (int off = 16; off >= 1; off >>= 1) q += __shfl_xor(q, off, 32);
  float rstd = rsqrtf(q * (1.0f / 128.0f) + 1e-5f);
  float4 g4 = *(const float4*)(gamma + lane * 4);
  float4 b4 = *(const float4*)(beta + lane * 4);
  float4 o;
  o.x = dx * rstd * g4.x + b4.x;  o.x = (o.x > 0.0f) ? o.x : 0.01f * o.x;
  o.y = dy * rstd * g4.y + b4.y;  o.y = (o.y > 0.0f) ? o.y : 0.01f * o.y;
  o.z = dz * rstd * g4.z + b4.z;  o.z = (o.z > 0.0f) ? o.z : 0.01f * o.z;
  o.w = dw * rstd * g4.w + b4.w;  o.w = (o.w > 0.0f) ? o.w : 0.01f * o.w;
  *(float4*)(out + (size_t)node * DIM + lane * 4) = o;
}

extern "C" void kernel_launch(void* const* d_in, const int* in_sizes, int n_in,
                              void* d_out, int out_size, void* d_ws, size_t ws_size,
                              hipStream_t stream) {
  (void)n_in; (void)out_size; (void)ws_size;
  const float* x      = (const float*)d_in[0];
  const int*   ei     = (const int*)d_in[1];
  const float* W_in   = (const float*)d_in[2];
  const float* b_in   = (const float*)d_in[3];
  const float* Wl     = (const float*)d_in[4];
  const float* att_s  = (const float*)d_in[5];
  const float* att_d  = (const float*)d_in[6];
  const float* bias_l = (const float*)d_in[7];
  const float* gamma  = (const float*)d_in[8];
  const float* beta   = (const float*)d_in[9];
  float* out = (float*)d_out;

  const int NN = in_sizes[0] / DIM;   // 100000
  const int EE = in_sizes[1] / 2;     // 1600000
  const int* src = ei;
  const int* dst = ei + EE;

  // Workspace carve-out (256B aligned)
  char* ws = (char*)d_ws;
  size_t off = 0;
  auto carve = [&](size_t bytes) -> char* {
    char* p = ws + off;
    off = (off + bytes + 255) & ~(size_t)255;
    return p;
  };
  float*    h_buf  = (float*)carve((size_t)NN * DIM * sizeof(float));
  float*    g_buf  = (float*)carve((size_t)NN * DIM * sizeof(float));
  float*    asrc_n = (float*)carve((size_t)NN * sizeof(float));
  float*    adst_n = (float*)carve((size_t)NN * sizeof(float));
  float*    nmax   = (float*)carve((size_t)NN * sizeof(float));
  float*    ndenom = (float*)carve((size_t)NN * sizeof(float));
  float*    ebuf   = (float*)carve((size_t)EE * sizeof(float));
  _Float16* Wpk    = (_Float16*)carve((size_t)5 * 16384 * sizeof(_Float16));

  const int mtiles = NN / 16;  // 6250 (N divisible by 16)

  // Pack all 5 weight matrices into WMMA B-fragment order
  {
    int total = 5 * 16384;
    pack_w_kernel<<<(total + 255) / 256, 256, 0, stream>>>(W_in, Wl, Wpk, total);
  }

  // Input layer: h = leaky(x @ W_in + b_in, 0.01)
  gemm_kernel<<<mtiles, 256, 0, stream>>>(x, Wpk, 0, h_buf, b_in, 1);

  for (int i = 0; i < NLAYERS; ++i) {
    // g = h @ Wl[i]
    gemm_kernel<<<mtiles, 256, 0, stream>>>(h_buf, Wpk, 1 + i, g_buf, nullptr, 0);
    // per-node attention logits
    alphas_kernel<<<(NN + 7) / 8, 256, 0, stream>>>(
        g_buf, att_s + i * DIM, att_d + i * DIM, asrc_n, adst_n, NN);
    // acc (reuses h_buf) = bias; nmax=-inf; ndenom=0
    init_kernel<<<(NN * DIM + 255) / 256, 256, 0, stream>>>(
        h_buf, bias_l + i * DIM, nmax, ndenom, NN);
    // edge softmax
    edge_max_kernel<<<(EE + 255) / 256, 256, 0, stream>>>(
        src, dst, asrc_n, adst_n, ebuf, nmax, EE);
    edge_exp_kernel<<<(EE + 255) / 256, 256, 0, stream>>>(
        dst, nmax, ebuf, ndenom, EE);
    // weighted scatter-add into acc
    scatter_kernel<<<(EE + 7) / 8, 256, 0, stream>>>(
        src, dst, ebuf, ndenom, g_buf, h_buf, EE);
    // LayerNorm + leaky; last layer writes to d_out
    ln_kernel<<<(NN + 7) / 8, 256, 0, stream>>>(
        h_buf, gamma + i * DIM, beta + i * DIM,
        (i == NLAYERS - 1) ? out : h_buf, NN);
  }
}
